// TransformerBlock_27874337751661
// MI455X (gfx1250) — matrix operand order repaired and checked by reading, not compiled, not run on hardware
//
#include <hip/hip_runtime.h>
#include <math.h>

// ---- problem constants (from the reference) ----
#define D_MODEL 1024
#define ROWS    4096      // B*L = 2*2048
#define INNER   4096
#define NQKV    3072
#define HEADS   16
#define HD      64
#define LSEQ    2048

typedef __attribute__((ext_vector_type(16))) __bf16 v16bf;
typedef __attribute__((ext_vector_type(8)))  __bf16 v8bf;
typedef __attribute__((ext_vector_type(8)))  float  v8f;
typedef unsigned int u32x4 __attribute__((ext_vector_type(4)));
typedef int          i32x4 __attribute__((ext_vector_type(4)));
typedef int          i32x8 __attribute__((ext_vector_type(8)));

__device__ __forceinline__ unsigned short f2bf(float f) {
  unsigned int u = __float_as_uint(f);
  u += 0x7fffu + ((u >> 16) & 1u);      // round-to-nearest-even
  return (unsigned short)(u >> 16);
}

// Build a 16-half fragment from two 16B LDS chunks (callers pass adjacent chunks, so
// lane half h holds k = 16h..16h+15, the same order for A and B).
__device__ __forceinline__ v16bf frag16(const unsigned short* p0, const unsigned short* p1) {
  v8bf lo = *(const v8bf*)p0;
  v8bf hi = *(const v8bf*)p1;
  return __builtin_shufflevector(lo, hi, 0,1,2,3,4,5,6,7,8,9,10,11,12,13,14,15);
}

__device__ __forceinline__ v8f wmma_bf16(v16bf a, v16bf b, v8f c) {
  // D(16x16,f32) = A(16x32,bf16) * B(32x16,bf16) + C
  return __builtin_amdgcn_wmma_f32_16x16x32_bf16(false, a, false, b, (short)0, c, false, false);
}

// ---- CDNA5 async copy: global -> LDS, tracked with ASYNCcnt ----
__device__ __forceinline__ void async_ld_b128(unsigned lds_off, const unsigned short* g) {
  unsigned long long ga = (unsigned long long)(uintptr_t)g;
  asm volatile("global_load_async_to_lds_b128 %0, %1, off"
               :: "v"(lds_off), "v"(ga) : "memory");
}
__device__ __forceinline__ void wait_async0() {
  asm volatile("s_wait_asynccnt 0x0" ::: "memory");
}

// ---- CDNA5 Tensor Data Mover: 2-D tile copy global -> LDS (TENSORcnt) ----
// tile: tile0 contiguous elements (2B) per row, tile1 rows, row stride strideElems.
// Optional LDS padding: pad_interval/pad_amount codes per D# group1 encoding
// (interval code 4 -> every 32 DWORDs, amount code 3 -> 4 DWORDs = 16B).
__device__ __forceinline__ void tdm_load_tile_2d(const unsigned short* gsrc,
                                                 unsigned ldsAddr,
                                                 int tile0, int tile1,
                                                 long long strideElems,
                                                 int dim0, int dim1,
                                                 int padICode, int padACode) {
  unsigned long long ga = (unsigned long long)(uintptr_t)gsrc;
  u32x4 g0;
  g0[0] = 1u;                                   // count=1, user descriptor
  g0[1] = ldsAddr;                              // lds_addr (bytes)
  g0[2] = (unsigned)ga;                         // global_addr[31:0]
  g0[3] = (unsigned)((ga >> 32) & 0x01FFFFFFull) | 0x80000000u;  // addr[56:32], type=2
  i32x8 g1;
  int w0 = (1 << 16);                           // data_size = 1 -> 2 bytes
  if (padACode >= 0) w0 |= (1 << 20) | (padICode << 22) | (padACode << 25);
  g1[0] = w0;
  g1[1] = (dim0 & 0xFFFF) << 16;                              // tensor_dim0[15:0]
  g1[2] = ((dim0 >> 16) & 0xFFFF) | ((dim1 & 0xFFFF) << 16);  // dim0 hi | dim1 lo
  g1[3] = ((dim1 >> 16) & 0xFFFF) | ((tile0 & 0xFFFF) << 16); // dim1 hi | tile_dim0
  g1[4] = (tile1 & 0xFFFF);                                   // tile_dim1 (tile_dim2=0)
  g1[5] = (int)(strideElems & 0xFFFFFFFFll);                  // tensor_dim0_stride lo
  g1[6] = (int)((strideElems >> 32) & 0xFFFFll);              // stride hi (dim1_stride=0)
  g1[7] = 0;
  const i32x4 z4 = (i32x4){0, 0, 0, 0};
#if __clang_major__ >= 23
  const i32x8 z8 = (i32x8){0, 0, 0, 0, 0, 0, 0, 0};
  __builtin_amdgcn_tensor_load_to_lds(g0, g1, z4, z4, z8, 0);
#else
  __builtin_amdgcn_tensor_load_to_lds(g0, g1, z4, z4, 0);
#endif
}

// ---------------------------------------------------------------------------
// fp32 -> bf16 elementwise conversion
// ---------------------------------------------------------------------------
__global__ __launch_bounds__(256) void f32_to_bf16_kernel(const float* __restrict__ in,
                                                          unsigned short* __restrict__ out,
                                                          int n) {
  int i = blockIdx.x * 256 + threadIdx.x;
  if (i < n) out[i] = f2bf(in[i]);
}

// ---------------------------------------------------------------------------
// LayerNorm over D_MODEL, bf16 output. One block per row.
// ---------------------------------------------------------------------------
__global__ __launch_bounds__(256) void ln_bf16_kernel(const float* __restrict__ x,
                                                      const float* __restrict__ g,
                                                      const float* __restrict__ beta,
                                                      unsigned short* __restrict__ out) {
  __shared__ float sred[8];
  const int row = blockIdx.x;
  const float* xr = x + (size_t)row * D_MODEL;

  float s = 0.f;
  for (int i = threadIdx.x; i < D_MODEL; i += 256) s += xr[i];
  for (int dd = 1; dd < 32; dd <<= 1) s += __shfl_xor(s, dd, 32);
  if ((threadIdx.x & 31) == 0) sred[threadIdx.x >> 5] = s;
  __syncthreads();
  float tot = 0.f;
  for (int w = 0; w < 8; ++w) tot += sred[w];
  const float mean = tot * (1.0f / (float)D_MODEL);
  __syncthreads();

  float v = 0.f;
  for (int i = threadIdx.x; i < D_MODEL; i += 256) { float d = xr[i] - mean; v += d * d; }
  for (int dd = 1; dd < 32; dd <<= 1) v += __shfl_xor(v, dd, 32);
  if ((threadIdx.x & 31) == 0) sred[threadIdx.x >> 5] = v;
  __syncthreads();
  float vt = 0.f;
  for (int w = 0; w < 8; ++w) vt += sred[w];
  const float rstd = rsqrtf(vt * (1.0f / (float)D_MODEL) + 1e-5f);

  for (int i = threadIdx.x; i < D_MODEL; i += 256)
    out[(size_t)row * D_MODEL + i] = f2bf((xr[i] - mean) * rstd * g[i] + beta[i]);
}

// ---------------------------------------------------------------------------
// Generic bf16 WMMA GEMM: C(M,N) = A(M,K) @ B(K,N) + bias [+ epilogue]
//   mode 0: bf16 out, q-columns (< D_MODEL) pre-scaled by hd^-0.5 (QKV)
//   mode 1: bf16 out, exact-erf GELU (FC1)
//   mode 2: f32 out = residual + acc + bias (out-proj / FC2)
// Block tile 128x128, K-step 64 (16 WMMAs per barrier pair). A tile staged by
// the Tensor Data Mover (wave 0 issues, pads rows 128B->+16B to give the 72-half
// LDS stride); B tile transposed via VGPR scatter.
// ---------------------------------------------------------------------------
__global__ __launch_bounds__(256) void gemm_bf16_kernel(
    const unsigned short* __restrict__ A,
    const unsigned short* __restrict__ B,
    const float* __restrict__ bias,
    const float* __restrict__ residual,
    void* __restrict__ out,
    int M, int N, int K, int mode)
{
  __shared__ unsigned short As[128][72];   // 128 x 64 (+8 pad), filled by TDM
  __shared__ unsigned short Bts[128][72];  // B^T tile: 128(N) x 64(K) (+8 pad)

  const int t    = threadIdx.x;
  const int lane = t & 31;
  const int lm   = lane & 15;
  const int wid  = t >> 5;
  const int wm   = (wid & 3) << 5;   // 0,32,64,96
  const int wn   = (wid >> 2) << 6;  // 0,64
  const int mBase = blockIdx.y << 7;
  const int nBase = blockIdx.x << 7;
  const int selA = (lane < 16) ? 0 : 16;
  const int selB = (lane < 16) ? 0 : 16;

  v8f acc[2][4];
#pragma unroll
  for (int i = 0; i < 2; ++i)
#pragma unroll
    for (int j = 0; j < 4; ++j)
      acc[i][j] = (v8f){0.f,0.f,0.f,0.f,0.f,0.f,0.f,0.f};

  const int brow = t >> 3;           // 0..31 (k, per pass)
  const int bcol = (t & 7) << 4;     // 0..112 step 16 (n)

  for (int k0 = 0; k0 < K; k0 += 64) {
    // A tile via TDM (one wave issues; pad 16B after each 128B row -> stride 72)
    if (wid == 0) {
      tdm_load_tile_2d(A + (size_t)mBase * K + k0, (unsigned)(uintptr_t)&As[0][0],
                       /*tile0*/64, /*tile1*/128, /*stride*/(long long)K,
                       /*dim0*/K, /*dim1*/M, /*padI(32dw)*/4, /*padA(4dw)*/3);
      __builtin_amdgcn_s_wait_tensorcnt(0);
    }

    // B tile: 2 passes, load 16 contiguous along N, scatter transposed
#pragma unroll
    for (int p = 0; p < 2; ++p) {
      const int kk = (p << 5) + brow;
      const uint4* bg = (const uint4*)(B + (size_t)(k0 + kk) * N + (nBase + bcol));
      union { uint4 v[2]; unsigned short u[16]; } bt;
      bt.v[0] = bg[0]; bt.v[1] = bg[1];
#pragma unroll
      for (int j = 0; j < 16; ++j) Bts[bcol + j][kk] = bt.u[j];
    }

    __syncthreads();

#pragma unroll
    for (int kc = 0; kc < 64; kc += 32) {
      v16bf af[2], bfr[4];
#pragma unroll
      for (int i = 0; i < 2; ++i) {
        const int r = wm + (i << 4) + lm;
        af[i] = frag16(&As[r][kc + selA], &As[r][kc + selA + 8]);
      }
#pragma unroll
      for (int j = 0; j < 4; ++j) {
        const int c = wn + (j << 4) + lm;
        bfr[j] = frag16(&Bts[c][kc + selB], &Bts[c][kc + selB + 8]);
      }
#pragma unroll
      for (int i = 0; i < 2; ++i)
#pragma unroll
        for (int j = 0; j < 4; ++j)
          acc[i][j] = wmma_bf16(af[i], bfr[j], acc[i][j]);
    }

    __syncthreads();
  }

  const int half8 = (lane >= 16) ? 8 : 0;
#pragma unroll
  for (int i = 0; i < 2; ++i) {
    const int rbase = mBase + wm + (i << 4) + half8;
#pragma unroll
    for (int j = 0; j < 4; ++j) {
      const int col = nBase + wn + (j << 4) + lm;
      const float bv = bias[col];
#pragma unroll
      for (int r = 0; r < 8; ++r) {
        const size_t idx = (size_t)(rbase + r) * N + col;
        float v = acc[i][j][r] + bv;
        if (mode == 0) {
          if (col < D_MODEL) v *= 0.125f;   // q * hd^-0.5
          ((unsigned short*)out)[idx] = f2bf(v);
        } else if (mode == 1) {
          v = 0.5f * v * (1.0f + erff(v * 0.70710678118654752f));
          ((unsigned short*)out)[idx] = f2bf(v);
        } else {
          ((float*)out)[idx] = residual[idx] + v;
        }
      }
    }
  }
}

// ---------------------------------------------------------------------------
// Causal flash attention. One block per (b, h, 128-query tile); 8 waves of 16
// query rows each; keys streamed in blocks of 64 with online softmax.
// Q/K tiles staged with global_load_async_to_lds_b128 (ASYNCcnt); V needs a
// transpose so it goes through a VGPR scatter. q pre-scaled by hd^-0.5.
// ---------------------------------------------------------------------------
__global__ __launch_bounds__(256) void attn_kernel(
    const unsigned short* __restrict__ qkv,   // [ROWS][NQKV] bf16
    unsigned short* __restrict__ obuf)        // [ROWS][D_MODEL] bf16
{
  __shared__ unsigned short Qs[128][72];      // 128 q-rows x 64 hd
  __shared__ unsigned short Ks[64][72];       // 64 keys x 64 hd (== B^T for QK^T)
  __shared__ unsigned short Vt[64][72];       // 64 hd x 64 keys (== B^T for PV)
  __shared__ unsigned short Ps[8][16][72];    // per-wave P: 16 q-rows x 64 keys

  const int t    = threadIdx.x;
  const int lane = t & 31;
  const int lm   = lane & 15;
  const int wid  = t >> 5;
  const int half8 = (lane >= 16) ? 8 : 0;
  const int selA  = (lane < 16) ? 0 : 16;
  const int selB  = (lane < 16) ? 0 : 16;

  const int qt = blockIdx.x & 15;
  const int h  = (blockIdx.x >> 4) & 15;
  const int bb = blockIdx.x >> 8;
  const int rowOff = bb * LSEQ;
  const int qBase  = qt << 7;

  // async-stage Q tile (completion folded into first iteration's wait)
  {
    const int r = t >> 2;         // 0..63
    const int c = (t & 3) << 4;   // 0..48
#pragma unroll
    for (int rr = 0; rr < 128; rr += 64) {
      const unsigned short* g = qkv + (size_t)(rowOff + qBase + rr + r) * NQKV + h * HD + c;
      async_ld_b128((unsigned)(uintptr_t)&Qs[rr + r][c],     g);
      async_ld_b128((unsigned)(uintptr_t)&Qs[rr + r][c + 8], g + 8);
    }
  }

  float mrow[8], lrow[8];
  v8f o[4];
#pragma unroll
  for (int r = 0; r < 8; ++r) { mrow[r] = -__builtin_inff(); lrow[r] = 0.f; }
#pragma unroll
  for (int nt = 0; nt < 4; ++nt) o[nt] = (v8f){0.f,0.f,0.f,0.f,0.f,0.f,0.f,0.f};

  const int ktEnd = 2 * qt + 1;   // last key block touching this query tile
  for (int kt = 0; kt <= ktEnd; ++kt) {
    const int kBase = kt << 6;
    __syncthreads();              // previous iteration done with Ks/Vt
    {
      const int r = t >> 2;         // key row 0..63
      const int c = (t & 3) << 4;   // hd chunk
      const unsigned short* gk = qkv + (size_t)(rowOff + kBase + r) * NQKV + D_MODEL + h * HD + c;
      async_ld_b128((unsigned)(uintptr_t)&Ks[r][c],     gk);
      async_ld_b128((unsigned)(uintptr_t)&Ks[r][c + 8], gk + 8);
      const uint4* gv = (const uint4*)(qkv + (size_t)(rowOff + kBase + r) * NQKV + 2 * D_MODEL + h * HD + c);
      union { uint4 v[2]; unsigned short u[16]; } vv;
      vv.v[0] = gv[0]; vv.v[1] = gv[1];
#pragma unroll
      for (int j = 0; j < 16; ++j) Vt[c + j][r] = vv.u[j];
    }
    wait_async0();                // Q (first iter) + K resident in LDS
    __syncthreads();

    // S = Q @ K^T  (wave's 16 rows x 64 keys)
    v8f s[4];
#pragma unroll
    for (int nt = 0; nt < 4; ++nt) s[nt] = (v8f){0.f,0.f,0.f,0.f,0.f,0.f,0.f,0.f};
#pragma unroll
    for (int k0 = 0; k0 < 64; k0 += 32) {
      const int qr = (wid << 4) + lm;
      v16bf aq = frag16(&Qs[qr][k0 + selA], &Qs[qr][k0 + selA + 8]);
#pragma unroll
      for (int nt = 0; nt < 4; ++nt) {
        v16bf bk = frag16(&Ks[(nt << 4) + lm][k0 + selB], &Ks[(nt << 4) + lm][k0 + selB + 8]);
        s[nt] = wmma_bf16(aq, bk, s[nt]);
      }
    }

    // causal mask (elementwise)
#pragma unroll
    for (int nt = 0; nt < 4; ++nt) {
      const int colg = kBase + (nt << 4) + lm;
#pragma unroll
      for (int r = 0; r < 8; ++r) {
        const int rowg = qBase + (wid << 4) + half8 + r;
        if (colg > rowg) s[nt][r] = -__builtin_inff();
      }
    }

    // online softmax; stash P (bf16) in the wave's private LDS slab
#pragma unroll
    for (int r = 0; r < 8; ++r) {
      float mx = s[0][r];
#pragma unroll
      for (int nt = 1; nt < 4; ++nt) mx = fmaxf(mx, s[nt][r]);
#pragma unroll
      for (int dd = 1; dd < 16; dd <<= 1) mx = fmaxf(mx, __shfl_xor(mx, dd, 32));
      const float mnew = fmaxf(mrow[r], mx);
      const float corr = __expf(mrow[r] - mnew);
      float sum = 0.f;
#pragma unroll
      for (int nt = 0; nt < 4; ++nt) {
        const float p = __expf(s[nt][r] - mnew);
        sum += p;
        Ps[wid][half8 + r][(nt << 4) + lm] = f2bf(p);
      }
#pragma unroll
      for (int dd = 1; dd < 16; dd <<= 1) sum += __shfl_xor(sum, dd, 32);
      lrow[r] = lrow[r] * corr + sum;
      mrow[r] = mnew;
#pragma unroll
      for (int nt = 0; nt < 4; ++nt) o[nt][r] *= corr;
    }

    // O += P @ V   (wave-private Ps; same-wave LDS ordering suffices)
#pragma unroll
    for (int kk = 0; kk < 64; kk += 32) {
      v16bf ap = frag16(&Ps[wid][lm][kk + selA], &Ps[wid][lm][kk + selA + 8]);
#pragma unroll
      for (int nt = 0; nt < 4; ++nt) {
        v16bf bv = frag16(&Vt[(nt << 4) + lm][kk + selB], &Vt[(nt << 4) + lm][kk + selB + 8]);
        o[nt] = wmma_bf16(ap, bv, o[nt]);
      }
    }
  }

  // normalize and store O as (b, l, h*hd) bf16
#pragma unroll
  for (int r = 0; r < 8; ++r) {
    const float inv = 1.0f / lrow[r];
    const int rowg = rowOff + qBase + (wid << 4) + half8 + r;
#pragma unroll
    for (int nt = 0; nt < 4; ++nt)
      obuf[(size_t)rowg * D_MODEL + h * HD + (nt << 4) + lm] = f2bf(o[nt][r] * inv);
  }
}

// ---------------------------------------------------------------------------
extern "C" void kernel_launch(void* const* d_in, const int* in_sizes, int n_in,
                              void* d_out, int out_size, void* d_ws, size_t ws_size,
                              hipStream_t stream) {
  (void)in_sizes; (void)n_in; (void)out_size; (void)ws_size;

  const float* x     = (const float*)d_in[0];
  const float* w_qkv = (const float*)d_in[1];
  const float* b_qkv = (const float*)d_in[2];
  const float* w_out = (const float*)d_in[3];
  const float* b_out = (const float*)d_in[4];
  const float* w_fc1 = (const float*)d_in[5];
  const float* b_fc1 = (const float*)d_in[6];
  const float* w_fc2 = (const float*)d_in[7];
  const float* b_fc2 = (const float*)d_in[8];
  const float* ln1_g = (const float*)d_in[9];
  const float* ln1_b = (const float*)d_in[10];
  const float* ln2_g = (const float*)d_in[11];
  const float* ln2_b = (const float*)d_in[12];

  size_t off = 0;
  auto take = [&](size_t bytes) {
    char* p = (char*)d_ws + off;
    off += (bytes + 255) & ~(size_t)255;
    return (void*)p;
  };
  unsigned short* wqkv = (unsigned short*)take((size_t)D_MODEL * NQKV * 2);
  unsigned short* wout = (unsigned short*)take((size_t)D_MODEL * D_MODEL * 2);
  unsigned short* wfc1 = (unsigned short*)take((size_t)D_MODEL * INNER * 2);
  unsigned short* wfc2 = (unsigned short*)take((size_t)INNER * D_MODEL * 2);
  unsigned short* hbf  = (unsigned short*)take((size_t)ROWS * D_MODEL * 2);
  unsigned short* qkvb = (unsigned short*)take((size_t)ROWS * NQKV * 2);
  unsigned short* obf  = (unsigned short*)take((size_t)ROWS * D_MODEL * 2);
  float*          x1   = (float*)take((size_t)ROWS * D_MODEL * 4);
  unsigned short* h2bf = (unsigned short*)take((size_t)ROWS * D_MODEL * 2);
  unsigned short* gbf  = (unsigned short*)take((size_t)ROWS * INNER * 2);

  auto cvt = [&](const float* src, unsigned short* dst, size_t n) {
    f32_to_bf16_kernel<<<dim3((unsigned)((n + 255) / 256)), dim3(256), 0, stream>>>(src, dst, (int)n);
  };
  cvt(w_qkv, wqkv, (size_t)D_MODEL * NQKV);
  cvt(w_out, wout, (size_t)D_MODEL * D_MODEL);
  cvt(w_fc1, wfc1, (size_t)D_MODEL * INNER);
  cvt(w_fc2, wfc2, (size_t)INNER * D_MODEL);

  // h = LN1(x) (bf16)
  ln_bf16_kernel<<<dim3(ROWS), dim3(256), 0, stream>>>(x, ln1_g, ln1_b, hbf);

  // qkv = h @ w_qkv + b_qkv  (q pre-scaled, bf16 out)
  gemm_bf16_kernel<<<dim3(NQKV / 128, ROWS / 128), dim3(256), 0, stream>>>(
      hbf, wqkv, b_qkv, nullptr, qkvb, ROWS, NQKV, D_MODEL, 0);

  // causal flash attention -> obf
  attn_kernel<<<dim3(2 * HEADS * (LSEQ / 128)), dim3(256), 0, stream>>>(qkvb, obf);

  // x1 = x + o @ w_out + b_out  (f32)
  gemm_bf16_kernel<<<dim3(D_MODEL / 128, ROWS / 128), dim3(256), 0, stream>>>(
      obf, wout, b_out, x, x1, ROWS, D_MODEL, D_MODEL, 2);

  // h2 = LN2(x1) (bf16)
  ln_bf16_kernel<<<dim3(ROWS), dim3(256), 0, stream>>>(x1, ln2_g, ln2_b, h2bf);

  // g = gelu(h2 @ w_fc1 + b_fc1) (bf16)
  gemm_bf16_kernel<<<dim3(INNER / 128, ROWS / 128), dim3(256), 0, stream>>>(
      h2bf, wfc1, b_fc1, nullptr, gbf, ROWS, INNER, D_MODEL, 1);

  // out = x1 + g @ w_fc2 + b_fc2 (f32)
  gemm_bf16_kernel<<<dim3(D_MODEL / 128, ROWS / 128), dim3(256), 0, stream>>>(
      gbf, wfc2, b_fc2, x1, (float*)d_out, ROWS, D_MODEL, INNER, 2);
}
